// GATActor_24215025615145
// MI455X (gfx1250) — matrix operand order repaired and checked by reading, not compiled, not run on hardware
//
#include <hip/hip_runtime.h>
#include <hip/hip_bf16.h>
#include <math.h>
#include <limits.h>

// ---------------- constants (match reference) ----------------
#define NNODES 20000
#define NPAD   20096        // 157 * 128
#define INF_K  131
#define K0PAD  160          // 131 padded to multiple of 32
#define HC     1024         // H*C
#define HHEADS 4
#define CDIM   256
#define OUTC   6
#define NEGSL  0.2f

typedef __bf16 bf16_t;
typedef __attribute__((ext_vector_type(16))) __bf16 v16bf;
typedef __attribute__((ext_vector_type(8)))  __bf16 v8bf;
typedef __attribute__((ext_vector_type(8)))  float  v8f;

static __device__ __forceinline__ v16bf make_frag(v8bf lo, v8bf hi) {
  return __builtin_shufflevector(lo, hi, 0,1,2,3,4,5,6,7,8,9,10,11,12,13,14,15);
}

static __device__ __forceinline__ float lrelu(float x) {
  return x >= 0.f ? x : NEGSL * x;
}

// monotonic int key for float atomicMax
static __device__ __forceinline__ int fkey(float f) {
  int b = __float_as_int(f);
  return b >= 0 ? b : (b ^ 0x7FFFFFFF);
}
static __device__ __forceinline__ float fdecode(int k) {
  int b = k >= 0 ? k : (k ^ 0x7FFFFFFF);
  return __int_as_float(b);
}

// ---------------- conversion kernels ----------------
__global__ void conv_act_kernel(const float* __restrict__ src, bf16_t* __restrict__ dst,
                                int rows, int K, int Kpad, unsigned total) {
  unsigned i = blockIdx.x * 256u + threadIdx.x;
  if (i >= total) return;
  unsigned r = i / (unsigned)Kpad;
  unsigned k = i % (unsigned)Kpad;
  float v = (r < (unsigned)rows && k < (unsigned)K) ? src[(size_t)r * K + k] : 0.0f;
  dst[i] = (bf16_t)v;
}

// W is [K, HC] row-major; produce Wt [HC, Kpad] (transposed, zero-padded K)
__global__ void conv_w_kernel(const float* __restrict__ W, bf16_t* __restrict__ Wt,
                              int K, int Kpad, unsigned total) {
  unsigned i = blockIdx.x * 256u + threadIdx.x;
  if (i >= total) return;
  unsigned n = i % (unsigned)HC;
  unsigned k = i / (unsigned)HC;
  float v = (k < (unsigned)K) ? W[(size_t)k * HC + n] : 0.0f;
  Wt[(size_t)n * Kpad + k] = (bf16_t)v;
}

// ---------------- WMMA bf16 GEMM: C[NPAD,HC] = A[NPAD,Kpad] * Wt[HC,Kpad]^T ----------------
__global__ __launch_bounds__(256) void gemm_bf16_wmma(const bf16_t* __restrict__ A,
                                                      const bf16_t* __restrict__ Bt,
                                                      float* __restrict__ C, int Kpad) {
  const int wid  = threadIdx.x >> 5;     // 0..7
  const int lane = threadIdx.x & 31;
  const int hl   = lane >> 4;            // half: 0 or 1
  const int l16  = lane & 15;
  const int m0 = blockIdx.x * 128 + (wid >> 1) * 32;   // wave: 32 rows
  const int n0 = blockIdx.y * 128 + (wid & 1) * 64;    // wave: 64 cols

  v8f acc[2][4];
#pragma unroll
  for (int ms = 0; ms < 2; ms++)
#pragma unroll
    for (int ns = 0; ns < 4; ns++) acc[ms][ns] = (v8f)(0.0f);

  for (int k = 0; k < Kpad; k += 32) {
    v16bf afrag[2];
#pragma unroll
    for (int ms = 0; ms < 2; ms++) {
      const bf16_t* ap = A + (size_t)(m0 + ms * 16 + l16) * Kpad + k;
      v8bf lo = *(const v8bf*)(ap + hl * 8);        // K = half*8 .. +7   -> comps 0..7
      v8bf hi = *(const v8bf*)(ap + 16 + hl * 8);   // K = 16+half*8..+7  -> comps 8..15
      afrag[ms] = make_frag(lo, hi);
    }
    v16bf bfrag[4];
#pragma unroll
    for (int ns = 0; ns < 4; ns++) {
      const bf16_t* bp = Bt + (size_t)(n0 + ns * 16 + l16) * Kpad + k;
      v8bf lo = *(const v8bf*)(bp + hl * 8);        // K = half*8 .. +7   -> comps 0..7
      v8bf hi = *(const v8bf*)(bp + 16 + hl * 8);   // K = 16+half*8..+7  -> comps 8..15
      bfrag[ns] = make_frag(lo, hi);
    }
#pragma unroll
    for (int ms = 0; ms < 2; ms++)
#pragma unroll
      for (int ns = 0; ns < 4; ns++)
        acc[ms][ns] = __builtin_amdgcn_wmma_f32_16x16x32_bf16(
            false, afrag[ms], false, bfrag[ns], (short)0, acc[ms][ns], false, false);
  }

  // C/D layout: lane l16 = col, vgpr v -> row = half*8 + v
#pragma unroll
  for (int ms = 0; ms < 2; ms++)
#pragma unroll
    for (int ns = 0; ns < 4; ns++) {
      int col = n0 + ns * 16 + l16;
#pragma unroll
      for (int v = 0; v < 8; v++) {
        int row = m0 + ms * 16 + hl * 8 + v;
        C[(size_t)row * HC + col] = acc[ms][ns][v];
      }
    }
}

// ---------------- per-node attention dots: a_s/a_d [N,H] ----------------
__global__ void attn_node_kernel(const float* __restrict__ hlin,
                                 const float* __restrict__ att_s,
                                 const float* __restrict__ att_d,
                                 float* __restrict__ a_s, float* __restrict__ a_d) {
  int wid  = threadIdx.x >> 5;
  int lane = threadIdx.x & 31;
  int pair = blockIdx.x * 8 + wid;               // pair = n*4 + h
  if (pair >= NNODES * HHEADS) return;
  int n = pair >> 2, h = pair & 3;
  size_t hb = (size_t)n * HC + h * CDIM + lane * 8;
  size_t ab = (size_t)h * CDIM + lane * 8;
  float4 h0 = *(const float4*)(hlin + hb);
  float4 h1 = *(const float4*)(hlin + hb + 4);
  float4 s0 = *(const float4*)(att_s + ab);
  float4 s1 = *(const float4*)(att_s + ab + 4);
  float4 d0 = *(const float4*)(att_d + ab);
  float4 d1 = *(const float4*)(att_d + ab + 4);
  float vs = h0.x*s0.x + h0.y*s0.y + h0.z*s0.z + h0.w*s0.w
           + h1.x*s1.x + h1.y*s1.y + h1.z*s1.z + h1.w*s1.w;
  float vd = h0.x*d0.x + h0.y*d0.y + h0.z*d0.z + h0.w*d0.w
           + h1.x*d1.x + h1.y*d1.y + h1.z*d1.z + h1.w*d1.w;
#pragma unroll
  for (int o = 16; o > 0; o >>= 1) {
    vs += __shfl_xor(vs, o, 32);
    vd += __shfl_xor(vd, o, 32);
  }
  if (lane == 0) { a_s[pair] = vs; a_d[pair] = vd; }
}

// ---------------- fills / inits ----------------
__global__ void fill_f32_kernel(float* __restrict__ p, float v, unsigned n) {
  unsigned i = blockIdx.x * 256u + threadIdx.x;
  if (i < n) p[i] = v;
}
__global__ void init_ms_kernel(int* __restrict__ mkey, float* __restrict__ ssum, unsigned n) {
  unsigned i = blockIdx.x * 256u + threadIdx.x;
  if (i < n) { mkey[i] = INT_MIN; ssum[i] = 0.0f; }
}

// ---------------- edge pass 1: logits + segment max ----------------
__global__ void edge_alpha_max_kernel(const float* __restrict__ a_s,
                                      const float* __restrict__ a_d,
                                      const int* __restrict__ srcIdx,
                                      const int* __restrict__ dstIdx,
                                      float* __restrict__ alpha,
                                      int* __restrict__ mkey, int E, int Etot) {
  int e = blockIdx.x * 256 + threadIdx.x;
  if (e >= Etot) return;
  int si, di;
  if (e < E) { si = srcIdx[e]; di = dstIdx[e]; } else { si = di = e - E; }
  float4 as = *(const float4*)(a_s + (size_t)si * 4);
  float4 ad = *(const float4*)(a_d + (size_t)di * 4);
  float4 al;
  al.x = lrelu(as.x + ad.x); al.y = lrelu(as.y + ad.y);
  al.z = lrelu(as.z + ad.z); al.w = lrelu(as.w + ad.w);
  *(float4*)(alpha + (size_t)e * 4) = al;
  int* mk = mkey + (size_t)di * 4;
  atomicMax(mk + 0, fkey(al.x));
  atomicMax(mk + 1, fkey(al.y));
  atomicMax(mk + 2, fkey(al.z));
  atomicMax(mk + 3, fkey(al.w));
}

// ---------------- edge pass 2: exp + segment sum ----------------
__global__ void edge_exp_sum_kernel(const int* __restrict__ srcIdx,
                                    const int* __restrict__ dstIdx,
                                    float* __restrict__ alpha,
                                    const int* __restrict__ mkey,
                                    float* __restrict__ ssum, int E, int Etot) {
  int e = blockIdx.x * 256 + threadIdx.x;
  if (e >= Etot) return;
  int di = (e < E) ? dstIdx[e] : (e - E);
  float4 al = *(float4*)(alpha + (size_t)e * 4);
  const int* mk = mkey + (size_t)di * 4;
  float4 ex;
  ex.x = expf(al.x - fdecode(mk[0]));
  ex.y = expf(al.y - fdecode(mk[1]));
  ex.z = expf(al.z - fdecode(mk[2]));
  ex.w = expf(al.w - fdecode(mk[3]));
  *(float4*)(alpha + (size_t)e * 4) = ex;   // overwrite with e values
  float* sp = ssum + (size_t)di * 4;
  atomicAdd(sp + 0, ex.x);
  atomicAdd(sp + 1, ex.y);
  atomicAdd(sp + 2, ex.z);
  atomicAdd(sp + 3, ex.w);
}

// ---------------- edge pass 3: weighted scatter-add aggregation ----------------
__global__ void aggregate_kernel(const float* __restrict__ hlin,
                                 const float* __restrict__ evals,
                                 const float* __restrict__ ssum,
                                 const int* __restrict__ srcIdx,
                                 const int* __restrict__ dstIdx,
                                 float* __restrict__ agg, int E) {
  int e = blockIdx.x;
  int si, di;
  if (e < E) { si = srcIdx[e]; di = dstIdx[e]; } else { si = di = e - E; }
  int t = threadIdx.x;
  int c = t * 4;                  // 256 threads * 4 = 1024
  int head = c >> 8;
  float ev = evals[(size_t)e * 4 + head];
  float sv = ssum[(size_t)di * 4 + head];
  float coef = ev / (sv + 1e-16f);
  float4 hv = *(const float4*)(hlin + (size_t)si * HC + c);
  float* ap = agg + (size_t)di * HC + c;
  atomicAdd(ap + 0, hv.x * coef);
  atomicAdd(ap + 1, hv.y * coef);
  atomicAdd(ap + 2, hv.z * coef);
  atomicAdd(ap + 3, hv.w * coef);
}

// ---------------- bias + relu ----------------
__global__ void bias_relu_kernel(float* __restrict__ agg, const float* __restrict__ b, unsigned n) {
  unsigned i = blockIdx.x * 256u + threadIdx.x;
  if (i >= n) return;
  unsigned c = i & (HC - 1);
  float v = agg[i] + b[c];
  agg[i] = v > 0.f ? v : 0.f;
}

// ---------------- final FC + softmax (one wave per node) ----------------
__global__ void fc_softmax_kernel(const float* __restrict__ act,
                                  const float* __restrict__ fcW,
                                  const float* __restrict__ fcb,
                                  float* __restrict__ out) {
  int wid  = threadIdx.x >> 5;
  int lane = threadIdx.x & 31;
  int node = blockIdx.x * 8 + wid;
  if (node >= NNODES) return;
  const float* hr = act + (size_t)node * HC;
  float s[OUTC] = {0.f, 0.f, 0.f, 0.f, 0.f, 0.f};
  for (int k = lane; k < HC; k += 32) {
    float hv = hr[k];
    const float* wr = fcW + (size_t)k * OUTC;
#pragma unroll
    for (int j = 0; j < OUTC; j++) s[j] += hv * wr[j];
  }
#pragma unroll
  for (int j = 0; j < OUTC; j++)
#pragma unroll
    for (int o = 16; o > 0; o >>= 1) s[j] += __shfl_xor(s[j], o, 32);
  if (lane == 0) {
    float lg[OUTC], mx = -3.0e38f;
#pragma unroll
    for (int j = 0; j < OUTC; j++) { lg[j] = s[j] + fcb[j]; mx = fmaxf(mx, lg[j]); }
    float den = 0.f, ex[OUTC];
#pragma unroll
    for (int j = 0; j < OUTC; j++) { ex[j] = expf(lg[j] - mx); den += ex[j]; }
#pragma unroll
    for (int j = 0; j < OUTC; j++) {
      out[(size_t)node * OUTC + j] = ex[j] / den;                       // softmax
      out[(size_t)NNODES * OUTC + (size_t)node * OUTC + j] = lg[j];     // logits
    }
  }
}

// ---------------- host launch ----------------
extern "C" void kernel_launch(void* const* d_in, const int* in_sizes, int n_in,
                              void* d_out, int out_size, void* d_ws, size_t ws_size,
                              hipStream_t stream) {
  const float* x     = (const float*)d_in[0];
  const int*   edges = (const int*)d_in[1];
  const int E    = in_sizes[1] / 2;
  const int Etot = E + NNODES;
  const int* srcIdx = edges;
  const int* dstIdx = edges + E;

  // workspace carve-up (all 256B aligned)
  char* ws = (char*)d_ws;
  const size_t HBYTES = (size_t)NPAD * HC * sizeof(float);      // 82,313,216
  float* buf0 = (float*)ws;              ws += HBYTES;          // act / agg
  float* buf1 = (float*)ws;              ws += HBYTES;          // hlin (GEMM out)
  bf16_t* Abf = (bf16_t*)ws;             ws += (size_t)NPAD * HC * 2;
  bf16_t* Wbf = (bf16_t*)ws;             ws += (size_t)HC * HC * 2;
  float* a_s  = (float*)ws;              ws += (size_t)NNODES * HHEADS * 4;
  float* a_d  = (float*)ws;              ws += (size_t)NNODES * HHEADS * 4;
  int*   mkey = (int*)ws;                ws += (size_t)NNODES * HHEADS * 4;
  float* ssum = (float*)ws;              ws += (size_t)NNODES * HHEADS * 4;
  float* alph = (float*)ws;              ws += (size_t)Etot * HHEADS * 4;

  const unsigned nNH   = NNODES * HHEADS;                 // 80000
  const unsigned nAGG  = (unsigned)((size_t)NPAD * HC);   // 20,578,304
  const unsigned nACT  = (unsigned)((size_t)NNODES * HC); // 20,480,000
  const int egrid = (Etot + 255) / 256;

  for (int l = 0; l < 3; l++) {
    const float* W    = (const float*)d_in[2 + 4 * l];
    const float* atts = (const float*)d_in[3 + 4 * l];
    const float* attd = (const float*)d_in[4 + 4 * l];
    const float* bias = (const float*)d_in[5 + 4 * l];
    const int K    = (l == 0) ? INF_K : HC;
    const int Kpad = (l == 0) ? K0PAD : HC;

    // 1) activations -> bf16 [NPAD, Kpad]
    {
      unsigned total = (unsigned)((size_t)NPAD * Kpad);
      const float* src = (l == 0) ? x : buf0;
      int rK = (l == 0) ? INF_K : HC;
      conv_act_kernel<<<(total + 255) / 256, 256, 0, stream>>>(src, Abf, NNODES, rK, Kpad, total);
    }
    // 2) weights -> bf16 transposed [HC, Kpad]
    {
      unsigned total = (unsigned)((size_t)HC * Kpad);
      conv_w_kernel<<<(total + 255) / 256, 256, 0, stream>>>(W, Wbf, K, Kpad, total);
    }
    // 3) GEMM: buf1 = A * Wt^T
    gemm_bf16_wmma<<<dim3(NPAD / 128, HC / 128), 256, 0, stream>>>(Abf, Wbf, buf1, Kpad);
    // 4) per-node attention dots
    attn_node_kernel<<<(nNH + 7) / 8, 256, 0, stream>>>(buf1, atts, attd, a_s, a_d);
    // 5) init segment max/sum + zero aggregation target
    init_ms_kernel<<<(nNH + 255) / 256, 256, 0, stream>>>(mkey, ssum, nNH);
    fill_f32_kernel<<<nAGG / 256, 256, 0, stream>>>(buf0, 0.0f, nAGG);
    // 6) edge logits + segment max
    edge_alpha_max_kernel<<<egrid, 256, 0, stream>>>(a_s, a_d, srcIdx, dstIdx, alph, mkey, E, Etot);
    // 7) exp + segment sum
    edge_exp_sum_kernel<<<egrid, 256, 0, stream>>>(srcIdx, dstIdx, alph, mkey, ssum, E, Etot);
    // 8) weighted scatter-add aggregation
    aggregate_kernel<<<Etot, 256, 0, stream>>>(buf1, alph, ssum, srcIdx, dstIdx, buf0, E);
    // 9) bias + relu (buf0 becomes next layer's activations)
    bias_relu_kernel<<<nACT / 256, 256, 0, stream>>>(buf0, bias, nACT);
  }

  const float* fcW = (const float*)d_in[14];
  const float* fcb = (const float*)d_in[15];
  fc_softmax_kernel<<<(NNODES + 7) / 8, 256, 0, stream>>>(buf0, fcW, fcb, (float*)d_out);
}
